// SSMPhonotacticsModel_21234318312115
// MI455X (gfx1250) — matrix operand order repaired and checked by reading, not compiled, not run on hardware
//
#include <hip/hip_runtime.h>

// CDNA5 / gfx1250, wave32. Uses v_wmma_f32_16x16x32_f16 for the fused
// logits GEMM + logsumexp. Operand layouts used here (contiguous k, same in A and B):
//  A (16x32 f16): lane L (M=L&15, hi=L>>4): frag half pair v holds
//    K = 16*hi + 2v .. +1
//  B (32x16 f16): lane L (N=L&15, hi=L>>4): frag half j holds K = 16*hi + j
//  C/D (16x16 f32): VGPR r, lane L: row M = r + 8*hi, col N = L&15

typedef __attribute__((ext_vector_type(16))) _Float16 v16h;
typedef __attribute__((ext_vector_type(8)))  float    v8f;

union H16 { v16h v; int4 q[2]; };

// sum across 16 lanes within each wave32 half via LDS-swizzle xor
#define SWZ_XOR_ADD(x, m) \
  ((x) + __int_as_float(__builtin_amdgcn_ds_swizzle(__float_as_int(x), 0x1f | ((m) << 10))))

__global__ __launch_bounds__(256) void cvtC_kernel(const float* __restrict__ C,
                                                   _Float16* __restrict__ Ch) {
  int i = blockIdx.x * 256 + threadIdx.x;   // 2048*64 = 131072 elements, grid 512
  Ch[i] = (_Float16)C[i];
}

template <bool F16C>
__global__ __launch_bounds__(256) void ssm_kernel(const int* __restrict__ xs,
                                                  const float* __restrict__ A,
                                                  const float* __restrict__ Bm,
                                                  const float* __restrict__ Cf,
                                                  const _Float16* __restrict__ Ch,
                                                  float* __restrict__ out) {
  constexpr int S = 2048, T = 512, STR = 68;  // STR padded: float4-aligned, few conflicts
  __shared__ float st_s[T * STR];             // pre-update states, f32
  __shared__ float tgt_s[T];                  // y[t, x_t]
  __shared__ float red[16];

  const int tid = threadIdx.x;
  const int seq = blockIdx.x;
  const int* xseq = xs + seq * T;

  // ---- Phase 1: serial recurrence s_{t+1} = A s_t + B[:,x_t], wave 0 only.
  // Wave32 lockstep + same-wave in-order LDS => no block barriers needed.
  if (tid < 32) {
    const int l = tid;
    float a0[64], a1[64];  // rows l and l+32 of A in registers
#pragma unroll
    for (int j = 0; j < 64; j += 4) {
      float4 r0 = *(const float4*)(A + l * 64 + j);
      float4 r1 = *(const float4*)(A + (l + 32) * 64 + j);
      a0[j] = r0.x; a0[j + 1] = r0.y; a0[j + 2] = r0.z; a0[j + 3] = r0.w;
      a1[j] = r1.x; a1[j + 1] = r1.y; a1[j + 2] = r1.z; a1[j + 3] = r1.w;
    }
    float cs0 = (l == 0) ? 1.0f : 0.0f;  // init = e0
    float cs1 = 0.0f;
    int x0 = xseq[0];
    float buc0 = Bm[l * S + x0];         // B column for step 0
    float buc1 = Bm[(l + 32) * S + x0];
#pragma unroll 1
    for (int t = 0; t < T; ++t) {
      st_s[t * STR + l] = cs0;           // emit PRE-update state
      st_s[t * STR + l + 32] = cs1;
      if (t == T - 1) break;
      __asm__ volatile("" ::: "memory");
      int xn = xseq[t + 1];              // prefetch next B column early
      float bun0 = Bm[l * S + xn];
      float bun1 = Bm[(l + 32) * S + xn];
      float acc0 = buc0, acc1 = buc1;
#pragma unroll
      for (int j = 0; j < 64; j += 4) {
        float4 sv = *(const float4*)(st_s + t * STR + j);  // broadcast read
        acc0 += a0[j] * sv.x + a0[j + 1] * sv.y + a0[j + 2] * sv.z + a0[j + 3] * sv.w;
        acc1 += a1[j] * sv.x + a1[j + 1] * sv.y + a1[j + 2] * sv.z + a1[j + 3] * sv.w;
      }
      __asm__ volatile("" ::: "memory");
      cs0 = acc0; cs1 = acc1;
      buc0 = bun0; buc1 = bun1;
    }
  }
  __syncthreads();

  // ---- Phase 1.5: target logits y[t, x_t] = states[t] . C[x_t], f32.
#pragma unroll
  for (int rep = 0; rep < 2; ++rep) {
    int t = tid + rep * 256;
    int xt = xseq[t];
    const float* crow = Cf + xt * 64;
    float acc = 0.0f;
#pragma unroll
    for (int j = 0; j < 64; j += 4) {
      float4 sv = *(const float4*)(st_s + t * STR + j);
      float4 cv = *(const float4*)(crow + j);
      acc += sv.x * cv.x + sv.y * cv.y + sv.z * cv.z + sv.w * cv.w;
    }
    tgt_s[t] = acc;
  }
  __syncthreads();

  // ---- Phase 2: logits GEMM (512x64)@(64x2048) fused with sum(exp(.)).
  // Wave w owns t-rows [w*64, w*64+64): 4 tiles of 16 rows; 128 s-tiles of 16 cols.
  const int w = tid >> 5;
  const int L = tid & 31;
  const int hi = L >> 4;
  const int M = L & 15;  // row-in-tile for A frag; col for B and C frags

  H16 af[4][2];  // A fragments: [t-tile][K-chunk(32)]
#pragma unroll
  for (int tt = 0; tt < 4; ++tt) {
    const int t0 = w * 64 + tt * 16;
#pragma unroll
    for (int c2 = 0; c2 < 2; ++c2) {
      const int k0 = 32 * c2;
#pragma unroll
      for (int v = 0; v < 8; ++v) {
        const int K = k0 + 16 * hi + 2 * v;
        float2 f = *(const float2*)(st_s + (t0 + M) * STR + K);
        af[tt][c2].v[2 * v]     = (_Float16)f.x;
        af[tt][c2].v[2 * v + 1] = (_Float16)f.y;
      }
    }
  }

  float lacc[4][8];  // per-lane running sum of exp over this lane's N-slice
#pragma unroll
  for (int tt = 0; tt < 4; ++tt)
#pragma unroll
    for (int r = 0; r < 8; ++r) lacc[tt][r] = 0.0f;

#pragma unroll 1
  for (int stile = 0; stile < 128; ++stile) {
    const int s0 = stile * 16;
    H16 bf[2];  // B fragments for K-chunks 0..31 / 32..63 (B[K][N] = C[s0+N][K])
    if (F16C) {
#pragma unroll
      for (int c2 = 0; c2 < 2; ++c2) {
        const int4* p = (const int4*)(Ch + (s0 + M) * 64 + 32 * c2 + 16 * hi);
        bf[c2].q[0] = p[0];
        bf[c2].q[1] = p[1];
      }
    } else {
#pragma unroll
      for (int c2 = 0; c2 < 2; ++c2) {
        const float* src = Cf + (s0 + M) * 64 + 32 * c2 + 16 * hi;
#pragma unroll
        for (int q4 = 0; q4 < 4; ++q4) {
          float4 f = *(const float4*)(src + 4 * q4);
          bf[c2].v[4 * q4]     = (_Float16)f.x;
          bf[c2].v[4 * q4 + 1] = (_Float16)f.y;
          bf[c2].v[4 * q4 + 2] = (_Float16)f.z;
          bf[c2].v[4 * q4 + 3] = (_Float16)f.w;
        }
      }
    }
#pragma unroll
    for (int tt = 0; tt < 4; ++tt) {
      v8f c = {};
      c = __builtin_amdgcn_wmma_f32_16x16x32_f16(false, af[tt][0].v, false, bf[0].v,
                                                 (short)0, c, false, false);
      c = __builtin_amdgcn_wmma_f32_16x16x32_f16(false, af[tt][1].v, false, bf[1].v,
                                                 (short)0, c, false, false);
      // |y| << 1 (inputs scaled 1/100) => exp cannot overflow; skip max-shift.
#pragma unroll
      for (int r = 0; r < 8; ++r) lacc[tt][r] += __expf(c[r]);
    }
  }

  // Combine the 16 lanes of each half (columns) -> per-row sum(exp); then LL.
  float ll = 0.0f;
#pragma unroll
  for (int tt = 0; tt < 4; ++tt) {
#pragma unroll
    for (int r = 0; r < 8; ++r) {
      float se = lacc[tt][r];
      se = SWZ_XOR_ADD(se, 1);
      se = SWZ_XOR_ADD(se, 2);
      se = SWZ_XOR_ADD(se, 4);
      se = SWZ_XOR_ADD(se, 8);
      const int t = w * 64 + tt * 16 + r + 8 * hi;
      ll += tgt_s[t] - __logf(se);
    }
  }
  if (M == 0) red[w * 2 + hi] = ll;  // lanes 0 and 16 of each wave
  __syncthreads();
  if (tid == 0) {
    float o = 0.0f;
#pragma unroll
    for (int k = 0; k < 16; ++k) o += red[k];
    out[seq] = o;
  }
}

extern "C" void kernel_launch(void* const* d_in, const int* in_sizes, int n_in,
                              void* d_out, int out_size, void* d_ws, size_t ws_size,
                              hipStream_t stream) {
  const int*   xs = (const int*)d_in[0];    // (64, 512) int32
  const float* A  = (const float*)d_in[1];  // (64, 64)
  const float* Bm = (const float*)d_in[2];  // (64, 2048)
  const float* C  = (const float*)d_in[3];  // (2048, 64)
  float* out = (float*)d_out;               // (64,)

  const size_t chBytes = (size_t)2048 * 64 * sizeof(_Float16);
  if (ws_size >= chBytes) {
    _Float16* Ch = (_Float16*)d_ws;
    cvtC_kernel<<<dim3(512), dim3(256), 0, stream>>>(C, Ch);
    ssm_kernel<true><<<dim3(64), dim3(256), 0, stream>>>(xs, A, Bm, C, Ch, out);
  } else {
    ssm_kernel<false><<<dim3(64), dim3(256), 0, stream>>>(xs, A, Bm, C, nullptr, out);
  }
}
